// SpikingParrot_45466523796143
// MI455X (gfx1250) — matrix operand order repaired and checked by reading, not compiled, not run on hardware
//
#include <hip/hip_runtime.h>

typedef __bf16 bf16;
typedef __attribute__((ext_vector_type(16))) __bf16 v16bf;
typedef __attribute__((ext_vector_type(8)))  __bf16 v8bf;
typedef __attribute__((ext_vector_type(8)))  float  v8f;

#if defined(__has_builtin)
#if __has_builtin(__builtin_amdgcn_tensor_load_to_lds) && \
    __has_builtin(__builtin_amdgcn_s_wait_tensorcnt)
#define HAVE_TDM 1
#endif
#endif
#ifndef HAVE_TDM
#define HAVE_TDM 0
#endif

#define WMMA_BF16(a, b, c) \
  __builtin_amdgcn_wmma_f32_16x16x32_bf16(false, (a), false, (b), (short)0, (c), false, false)

__device__ __forceinline__ v8f vzero8() {
  v8f v;
#pragma unroll
  for (int i = 0; i < 8; ++i) v[i] = 0.0f;
  return v;
}

__device__ __forceinline__ float sigmoidf(float x) { return 1.0f / (1.0f + __expf(-x)); }

#if HAVE_TDM
typedef unsigned int u32x4 __attribute__((ext_vector_type(4)));
typedef int i32x4 __attribute__((ext_vector_type(4)));
typedef int i32x8 __attribute__((ext_vector_type(8)));

// TDM 2D copy: `rows` lines of row_u8*8 bytes, global line stride stride_u8*8
// bytes, packed contiguously into LDS at byte offset lds_off. D# per CDNA5 ISA
// ch.8: group0 = {count=1, lds_addr, global_addr[56:0], type=2}; group1 =
// {data_size=3(8B), tensor_dim0=row_u8, tensor_dim1=rows, tile_dim0=row_u8,
// tile_dim1=rows, dim0_stride=stride_u8}; groups 2/3 zero (<=2D tensor).
__device__ __forceinline__ void tdm_copy2d(unsigned lds_off, const void* gsrc, unsigned rows,
                                           unsigned row_u8, unsigned long long stride_u8) {
  unsigned long long ga = (unsigned long long)gsrc;
  u32x4 g0;
  g0[0] = 1u;
  g0[1] = lds_off;
  g0[2] = (unsigned)ga;
  g0[3] = (unsigned)((ga >> 32) & 0x01ffffffu) | 0x80000000u;  // addr[56:32] | type=2
  i32x8 g1;
  g1[0] = (int)(3u << 16);                                                // data_size = 8B
  g1[1] = (int)((row_u8 & 0xffffu) << 16);                                // tensor_dim0 lo
  g1[2] = (int)(((row_u8 >> 16) & 0xffffu) | ((rows & 0xffffu) << 16));   // dim0 hi | dim1 lo
  g1[3] = (int)(((rows >> 16) & 0xffffu) | ((row_u8 & 0xffffu) << 16));   // dim1 hi | tile_dim0
  g1[4] = (int)(rows & 0xffffu);                                          // tile_dim1 (dim2=0)
  g1[5] = (int)(stride_u8 & 0xffffffffu);                                 // dim0_stride lo
  g1[6] = (int)((stride_u8 >> 32) & 0xffffu);                             // dim0_stride hi
  g1[7] = 0;
  i32x4 z4 = {0, 0, 0, 0};
  i32x8 z8 = {0, 0, 0, 0, 0, 0, 0, 0};
  __builtin_amdgcn_tensor_load_to_lds(g0, g1, z4, z4, z8, 0);
}
#endif

// ---- WMMA fragment loaders (wave32, 16x16x32 bf16) -------------------------
__device__ __forceinline__ v16bf a_frag_f32(const float* A, int lda, int k0, int lane) {
  int r  = lane & 15;
  int kb = k0 + ((lane >> 4) << 3);
  const float* p = A + (size_t)r * lda + kb;
  float4 x0 = *(const float4*)(p);
  float4 x1 = *(const float4*)(p + 4);
  float4 y0 = *(const float4*)(p + 16);
  float4 y1 = *(const float4*)(p + 20);
  v16bf v;
  v[0] = (bf16)x0.x; v[1] = (bf16)x0.y; v[2] = (bf16)x0.z; v[3] = (bf16)x0.w;
  v[4] = (bf16)x1.x; v[5] = (bf16)x1.y; v[6] = (bf16)x1.z; v[7] = (bf16)x1.w;
  v[8] = (bf16)y0.x; v[9] = (bf16)y0.y; v[10] = (bf16)y0.z; v[11] = (bf16)y0.w;
  v[12] = (bf16)y1.x; v[13] = (bf16)y1.y; v[14] = (bf16)y1.z; v[15] = (bf16)y1.w;
  return v;
}

__device__ __forceinline__ v16bf a_frag_bf16(const bf16* A, int lda, int k0, int lane) {
  int r  = lane & 15;
  int kb = k0 + ((lane >> 4) << 3);
  const bf16* p = A + (size_t)r * lda + kb;
  v8bf lo = *(const v8bf*)(p);
  v8bf hi = *(const v8bf*)(p + 16);
  return __builtin_shufflevector(lo, hi, 0, 1, 2, 3, 4, 5, 6, 7, 8, 9, 10, 11, 12, 13, 14, 15);
}

// Packed B layout: [ntile][ktile][lane(n=l%16)][16 k-values], 1KB per 32x16 tile
__device__ __forceinline__ v16bf b_frag_packed(const bf16* Wp, int ldk, int nt, int kt, int lane) {
  return *(const v16bf*)(Wp + (((size_t)nt * ldk + kt) * 32 + lane) * 16);
}

// ---- Weight pre-pack: W[N,K] fp32 -> bf16 WMMA-B-fragment layout -----------
__global__ void k_pack_w(const float* __restrict__ W, bf16* __restrict__ out, int N, int K) {
  int idx = blockIdx.x * blockDim.x + threadIdx.x;
  int ldk = K >> 5;
  int total = (N >> 4) * ldk * 32;
  if (idx >= total) return;
  int l = idx & 31;
  int tile = idx >> 5;
  int kt = tile % ldk, nt = tile / ldk;
  const float* p = W + (size_t)((nt << 4) + (l & 15)) * K + (kt << 5) + ((l >> 4) << 3);
  v16bf v;
#pragma unroll
  for (int s = 0; s < 16; ++s) v[s] = (bf16)p[(s & 7) + ((s >> 3) << 4)];
  *(v16bf*)(out + (size_t)idx * 16) = v;
}

// ---- Embedding gather + DyT ------------------------------------------------
__global__ void k_embed_dyt(const float* __restrict__ emb, const int* __restrict__ tok,
                            const float* __restrict__ alpha, const float* __restrict__ gamma,
                            const float* __restrict__ beta, float* __restrict__ outF,
                            float* __restrict__ outR, int S, int E, int total) {
  int i = blockIdx.x * blockDim.x + threadIdx.x;
  if (i >= total) return;
  int e = i % E;
  int bs = i / E;
  int s = bs % S;
  int b = bs / S;
  float v = emb[(size_t)tok[bs] * E + e];
  v = gamma[e] * tanhf(alpha[0] * v) + beta[e];
  outF[i] = v;
  if (outR) outR[((size_t)b * S + (S - 1 - s)) * E + e] = v;
}

// ---- GEMM: C[M,N] = A[M,K] @ W[N,K]^T + bias (W packed bf16) ---------------
// 128 threads = 4 waves; wave tile (M2?32:16) x 64. Each wave TDM-stages its
// own 4KB B panel per K-tile into LDS, double buffered (per-wave TENSORcnt).
template <bool M2>
__global__ __launch_bounds__(128) void k_gemm(const float* __restrict__ A,
                                              const bf16* __restrict__ Wp,
                                              const float* __restrict__ bias,
                                              float* __restrict__ C, int M, int N, int K) {
  constexpr int MI = M2 ? 2 : 1;
  int wave = threadIdx.x >> 5, lane = threadIdx.x & 31;
  int m0 = blockIdx.x * (M2 ? 32 : 16);
  int n0 = blockIdx.y * 256 + wave * 64;
  int ldk = K >> 5;
  int nt0 = n0 >> 4;
  v8f acc[MI][4];
#pragma unroll
  for (int mi = 0; mi < MI; ++mi)
#pragma unroll
    for (int j = 0; j < 4; ++j) acc[mi][j] = vzero8();

#if HAVE_TDM
  __shared__ bf16 sB[2][4][4 * 512];  // [buf][wave][j*512 + lane*16] = 32KB
  unsigned lds0 = (unsigned)(size_t)&sB[0][wave][0];
  unsigned lds1 = (unsigned)(size_t)&sB[1][wave][0];
  const bf16* gB = Wp + (size_t)nt0 * ldk * 512;  // tile (nt0+j, kt) at +(j*ldk+kt)*512
  unsigned long long strd = (unsigned long long)ldk * 128;
  tdm_copy2d(lds0, gB, 4, 128, strd);
  for (int kt = 0; kt < ldk; ++kt) {
    int buf = kt & 1;
    if (kt + 1 < ldk) {
      tdm_copy2d(buf ? lds0 : lds1, gB + (size_t)(kt + 1) * 512, 4, 128, strd);
      __builtin_amdgcn_s_wait_tensorcnt(1);
    } else {
      __builtin_amdgcn_s_wait_tensorcnt(0);
    }
    v16bf a[MI];
#pragma unroll
    for (int mi = 0; mi < MI; ++mi)
      a[mi] = a_frag_f32(A + (size_t)(m0 + mi * 16) * K, K, kt << 5, lane);
#pragma unroll
    for (int j = 0; j < 4; ++j) {
      v16bf b = *(const v16bf*)&sB[buf][wave][j * 512 + lane * 16];
#pragma unroll
      for (int mi = 0; mi < MI; ++mi) acc[mi][j] = WMMA_BF16(a[mi], b, acc[mi][j]);
    }
  }
#else
  for (int kt = 0; kt < ldk; ++kt) {
    v16bf a[MI];
#pragma unroll
    for (int mi = 0; mi < MI; ++mi)
      a[mi] = a_frag_f32(A + (size_t)(m0 + mi * 16) * K, K, kt << 5, lane);
    if (kt + 1 < ldk)
      __builtin_prefetch(Wp + (((size_t)nt0 * ldk + kt + 1) * 32 + lane) * 16, 0, 1);
#pragma unroll
    for (int j = 0; j < 4; ++j) {
      v16bf b = b_frag_packed(Wp, ldk, nt0 + j, kt, lane);
#pragma unroll
      for (int mi = 0; mi < MI; ++mi) acc[mi][j] = WMMA_BF16(a[mi], b, acc[mi][j]);
    }
  }
#endif
  int rb = (lane >> 4) << 3, cc = lane & 15;
#pragma unroll
  for (int j = 0; j < 4; ++j) {
    int c = n0 + j * 16 + cc;
    float bv = bias[c];
#pragma unroll
    for (int mi = 0; mi < MI; ++mi)
#pragma unroll
      for (int r = 0; r < 8; ++r)
        C[(size_t)(m0 + mi * 16 + rb + r) * N + c] = acc[mi][j][r] + bv;
  }
}

// ---- Grid barrier (agent-scope atomics; counter zeroed before launch) ------
__device__ __forceinline__ void gbar(int* bar, int nblk, int ph) {
  __syncthreads();
  if (threadIdx.x == 0) {
    __threadfence();
    atomicAdd(bar, 1);
    while (__hip_atomic_load(bar, __ATOMIC_ACQUIRE, __HIP_MEMORY_SCOPE_AGENT) < ph * nblk) {
      __builtin_amdgcn_s_sleep(1);
    }
    __threadfence();
  }
  __syncthreads();
}

// ---- sLSTM sequential scan -------------------------------------------------
// grid = H/64 blocks, 128 threads (4 waves). Block owns 64 hidden units; wave g
// computes gate type g as a 16x64 WMMA slice (K=H). syn/mem fp32 in LDS;
// mem/spk broadcast via bf16 L2 scratch, parity double-buffered, 1 grid
// barrier per step. Whh/Wfc panels TDM-staged into LDS, double buffered.
template <int H>
__global__ __launch_bounds__(128) void k_slstm_scan(
    const float* __restrict__ xg, const bf16* __restrict__ Whhp, const float* __restrict__ bhh,
    const bf16* __restrict__ Wfcp, const float* __restrict__ bfc,
    const float* __restrict__ thr_p, const float* __restrict__ alpha_p,
    const float* __restrict__ gamma, const float* __restrict__ beta,
    const float* __restrict__ x_res, float* __restrict__ out, const float* __restrict__ syn0,
    const float* __restrict__ mem0, int ld_in, float* __restrict__ synF, float* __restrict__ memF,
    int ld_out, bf16* __restrict__ membuf, bf16* __restrict__ spkbuf, int* __restrict__ bar,
    int S) {
  constexpr int NB = H / 64;
  constexpr int LDK = H / 32;
  const int j0 = blockIdx.x * 64;
  const int tid = threadIdx.x, wave = tid >> 5, lane = tid & 31;
  const float thr = thr_p[0], alpha = alpha_p[0];
  __shared__ float s_syn[16 * 64], s_mem[16 * 64], s_g[4][16 * 64];
#if HAVE_TDM
  __shared__ bf16 sW[2][4][4 * 512];  // weight staging, 32KB
  unsigned ldsw0 = (unsigned)(size_t)&sW[0][wave][0];
  unsigned ldsw1 = (unsigned)(size_t)&sW[1][wave][0];
#endif
  int ph = 0;

  for (int i = tid; i < 16 * 64; i += 128) {
    int b = i >> 6, n = i & 63;
    float sv = syn0 ? syn0[(size_t)b * ld_in + j0 + n] : 0.0f;
    float mv = mem0 ? mem0[(size_t)b * ld_in + j0 + n] : 0.0f;
    s_syn[i] = sv;
    s_mem[i] = mv;
    membuf[(size_t)b * H + j0 + n] = (bf16)mv;
  }
  gbar(bar, NB, ++ph);

  for (int t = 0; t < S; ++t) {
    const bf16* mprev = membuf + (size_t)(t & 1) * 16 * H;
    {  // gates
      v8f g0 = vzero8(), g1 = vzero8(), g2 = vzero8(), g3 = vzero8();
      const int nt0 = (wave * H + j0) >> 4;
#if HAVE_TDM
      const bf16* gW = Whhp + (size_t)nt0 * LDK * 512;
      tdm_copy2d(ldsw0, gW, 4, 128, (unsigned long long)LDK * 128);
      for (int kt = 0; kt < LDK; ++kt) {
        int buf = kt & 1;
        if (kt + 1 < LDK) {
          tdm_copy2d(buf ? ldsw0 : ldsw1, gW + (size_t)(kt + 1) * 512, 4, 128,
                     (unsigned long long)LDK * 128);
          __builtin_amdgcn_s_wait_tensorcnt(1);
        } else {
          __builtin_amdgcn_s_wait_tensorcnt(0);
        }
        v16bf a = a_frag_bf16(mprev, H, kt << 5, lane);
        g0 = WMMA_BF16(a, *(const v16bf*)&sW[buf][wave][0 * 512 + lane * 16], g0);
        g1 = WMMA_BF16(a, *(const v16bf*)&sW[buf][wave][1 * 512 + lane * 16], g1);
        g2 = WMMA_BF16(a, *(const v16bf*)&sW[buf][wave][2 * 512 + lane * 16], g2);
        g3 = WMMA_BF16(a, *(const v16bf*)&sW[buf][wave][3 * 512 + lane * 16], g3);
      }
#else
      for (int kt = 0; kt < LDK; ++kt) {
        v16bf a = a_frag_bf16(mprev, H, kt << 5, lane);
        g0 = WMMA_BF16(a, b_frag_packed(Whhp, LDK, nt0 + 0, kt, lane), g0);
        g1 = WMMA_BF16(a, b_frag_packed(Whhp, LDK, nt0 + 1, kt, lane), g1);
        g2 = WMMA_BF16(a, b_frag_packed(Whhp, LDK, nt0 + 2, kt, lane), g2);
        g3 = WMMA_BF16(a, b_frag_packed(Whhp, LDK, nt0 + 3, kt, lane), g3);
      }
#endif
      v8f accs[4] = {g0, g1, g2, g3};
      int rb = (lane >> 4) << 3, cc = lane & 15;
#pragma unroll
      for (int nt = 0; nt < 4; ++nt)
#pragma unroll
        for (int r = 0; r < 8; ++r) {
          int b = rb + r, n = nt * 16 + cc;
          float g = accs[nt][r] + xg[((size_t)b * S + t) * (4 * H) + wave * H + j0 + n] +
                    bhh[wave * H + j0 + n];
          s_g[wave][b * 64 + n] = g;
        }
    }
    __syncthreads();
    for (int i = tid; i < 16 * 64; i += 128) {
      int b = i >> 6, n = i & 63;
      float gi = s_g[0][i], gf = s_g[1][i], gg = s_g[2][i], go = s_g[3][i];
      float syn = s_syn[i], mem = s_mem[i];
      float reset = mem > thr ? 1.0f : 0.0f;
      float sn = sigmoidf(gf) * syn + sigmoidf(gi) * tanhf(gg);
      float h = sigmoidf(go) * tanhf(sn);
      float mn = h - reset * thr;
      float spike = (mn - thr) > 0.0f ? 1.0f : 0.0f;
      s_syn[i] = sn;
      s_mem[i] = mn;
      membuf[(size_t)((t + 1) & 1) * 16 * H + (size_t)b * H + j0 + n] = (bf16)mn;
      spkbuf[(size_t)(t & 1) * 16 * H + (size_t)b * H + j0 + n] = (bf16)spike;
    }
    gbar(bar, NB, ++ph);  // mem(t)/spk(t) visible everywhere
    {  // out = spk @ Wfc^T + bfc (+res), DyT
      const bf16* spk = spkbuf + (size_t)(t & 1) * 16 * H;
      v8f acc = vzero8();
      int c0 = j0 + wave * 16;
      int ntc = c0 >> 4;
#if HAVE_TDM
      const bf16* gF = Wfcp + (size_t)ntc * LDK * 512;
      tdm_copy2d(ldsw0, gF, 1, 128, 128);
      for (int kt = 0; kt < LDK; ++kt) {
        int buf = kt & 1;
        if (kt + 1 < LDK) {
          tdm_copy2d(buf ? ldsw0 : ldsw1, gF + (size_t)(kt + 1) * 512, 1, 128, 128);
          __builtin_amdgcn_s_wait_tensorcnt(1);
        } else {
          __builtin_amdgcn_s_wait_tensorcnt(0);
        }
        v16bf a = a_frag_bf16(spk, H, kt << 5, lane);
        acc = WMMA_BF16(a, *(const v16bf*)&sW[buf][wave][lane * 16], acc);
      }
#else
      for (int kt = 0; kt < LDK; ++kt) {
        v16bf a = a_frag_bf16(spk, H, kt << 5, lane);
        acc = WMMA_BF16(a, b_frag_packed(Wfcp, LDK, ntc, kt, lane), acc);
      }
#endif
      int rb = (lane >> 4) << 3, c = c0 + (lane & 15);
#pragma unroll
      for (int r = 0; r < 8; ++r) {
        int b = rb + r;
        float v = acc[r] + bfc[c];
        if (x_res) v += x_res[((size_t)b * S + t) * H + c];
        v = gamma[c] * tanhf(alpha * v) + beta[c];
        out[((size_t)b * S + t) * H + c] = v;
      }
    }
  }
  __syncthreads();
  for (int i = tid; i < 16 * 64; i += 128) {
    int b = i >> 6, n = i & 63;
    synF[(size_t)b * ld_out + j0 + n] = s_syn[i];
    memF[(size_t)b * ld_out + j0 + n] = s_mem[i];
  }
}

// ---- Host orchestration ----------------------------------------------------
// Flattened input order (pytree: sorted dict keys, depth-first):
//   dec[0]:  0 Wfc, 1 Whh, 2 Wih, 3 alpha, 4 beta, 5 bfc, 6 bhh, 7 bih, 8 gamma, 9 thr
//   dec[1]: 10..19 | enc[0].b: 20..29  enc[0].f: 30..39 | enc[1].b: 40..49  enc[1].f: 50..59
//   60 mem_W 61 mem_b 62 out_W 63 out_b 64 src_dyt.a 65 .beta 66 .gamma 67 src_emb
//   68 syn_W 69 syn_b 70 tgt_dyt.a 71 .beta 72 .gamma 73 tgt_emb 74 src 75 tgt
extern "C" void kernel_launch(void* const* d_in, const int* in_sizes, int n_in, void* d_out,
                              int out_size, void* d_ws, size_t ws_size, hipStream_t stream) {
  (void)in_sizes; (void)n_in; (void)out_size; (void)ws_size;
  constexpr int B = 16, S = 256, EMB = 512, HID = 512, H2 = 1024, V = 32000;
  auto F = [&](int i) { return (const float*)d_in[i]; };
  const int* src = (const int*)d_in[74];
  const int* tgt = (const int*)d_in[75];
  float* logits = (float*)d_out;

  char* wsp = (char*)d_ws;
  auto alloc = [&](size_t bytes) -> void* {
    void* p = wsp;
    wsp += (bytes + 255) & ~(size_t)255;
    return p;
  };

  auto pack = [&](const float* W, int N, int K) -> bf16* {
    bf16* p = (bf16*)alloc((size_t)N * K * 2);
    int total = (N * K) >> 4;
    k_pack_w<<<(total + 255) / 256, 256, 0, stream>>>(W, p, N, K);
    return p;
  };
  bf16 *pWih_e[2][2], *pWhh_e[2][2], *pWfc_e[2][2];
  const int encBase[2][2] = {{30, 20}, {50, 40}};
  for (int l = 0; l < 2; ++l)
    for (int d = 0; d < 2; ++d) {
      int bse = encBase[l][d];
      pWih_e[l][d] = pack(F(bse + 2), 4 * HID, EMB);
      pWhh_e[l][d] = pack(F(bse + 1), 4 * HID, HID);
      pWfc_e[l][d] = pack(F(bse + 0), HID, HID);
    }
  bf16* pWih_d0 = pack(F(2), 4 * H2, EMB);
  bf16* pWhh_d0 = pack(F(1), 4 * H2, H2);
  bf16* pWfc_d0 = pack(F(0), H2, H2);
  bf16* pWih_d1 = pack(F(12), 4 * H2, H2);
  bf16* pWhh_d1 = pack(F(11), 4 * H2, H2);
  bf16* pWfc_d1 = pack(F(10), H2, H2);
  bf16* pSynW = pack(F(68), H2, H2);
  bf16* pMemW = pack(F(60), H2, H2);
  bf16* pOutW = pack(F(62), V, H2);

  float* xg = (float*)alloc((size_t)B * S * 4 * H2 * 4);
  float* eF = (float*)alloc((size_t)B * S * EMB * 4);
  float* eB = (float*)alloc((size_t)B * S * EMB * 4);
  float* aF = (float*)alloc((size_t)B * S * HID * 4);
  float* aB = (float*)alloc((size_t)B * S * HID * 4);
  float* dIn = (float*)alloc((size_t)B * S * EMB * 4);
  float* dA1 = (float*)alloc((size_t)B * S * H2 * 4);
  float* dA2 = (float*)alloc((size_t)B * S * H2 * 4);
  float* catS = (float*)alloc((size_t)B * H2 * 4);
  float* catM = (float*)alloc((size_t)B * H2 * 4);
  float* synP = (float*)alloc((size_t)B * H2 * 4);
  float* memP = (float*)alloc((size_t)B * H2 * 4);
  float* dmpS = (float*)alloc((size_t)B * H2 * 4);
  float* dmpM = (float*)alloc((size_t)B * H2 * 4);
  bf16* membuf = (bf16*)alloc((size_t)2 * B * H2 * 2);
  bf16* spkbuf = (bf16*)alloc((size_t)2 * B * H2 * 2);
  int* bar = (int*)alloc(256);

  dim3 eg((B * S * EMB + 255) / 256);
  k_embed_dyt<<<eg, 256, 0, stream>>>(F(67), src, F(64), F(66), F(65), eF, eB, S, EMB, B * S * EMB);
  k_embed_dyt<<<eg, 256, 0, stream>>>(F(73), tgt, F(70), F(72), F(71), dIn, nullptr, S, EMB,
                                      B * S * EMB);

  auto gemm = [&](const float* A, const bf16* Wp, const float* bias, float* C, int M, int N,
                  int K) {
    if (M >= 32)
      k_gemm<true><<<dim3(M / 32, N / 256), 128, 0, stream>>>(A, Wp, bias, C, M, N, K);
    else
      k_gemm<false><<<dim3(1, N / 256), 128, 0, stream>>>(A, Wp, bias, C, M, N, K);
  };

  auto runEnc = [&](int l, int d, const float* A, const float* res, float* outb, float* sF,
                    float* mF) {
    int bse = encBase[l][d];
    gemm(A, pWih_e[l][d], F(bse + 7), xg, B * S, 4 * HID, EMB);
    (void)hipMemsetAsync(bar, 0, 64, stream);
    k_slstm_scan<HID><<<HID / 64, 128, 0, stream>>>(
        xg, pWhh_e[l][d], F(bse + 6), pWfc_e[l][d], F(bse + 5), F(bse + 9), F(bse + 3),
        F(bse + 8), F(bse + 4), res, outb, nullptr, nullptr, HID, sF, mF, H2, membuf, spkbuf, bar,
        S);
  };

  runEnc(0, 0, eF, nullptr, aF, dmpS, dmpM);
  runEnc(0, 1, eB, nullptr, aB, dmpS, dmpM);
  runEnc(1, 0, aF, aF, eF, catS + 0, catM + 0);
  runEnc(1, 1, aB, aB, eB, catS + 512, catM + 512);

  gemm(catS, pSynW, F(69), synP, 16, H2, H2);
  gemm(catM, pMemW, F(61), memP, 16, H2, H2);

  gemm(dIn, pWih_d0, F(7), xg, B * S, 4 * H2, EMB);
  (void)hipMemsetAsync(bar, 0, 64, stream);
  k_slstm_scan<H2><<<H2 / 64, 128, 0, stream>>>(xg, pWhh_d0, F(6), pWfc_d0, F(5), F(9), F(3), F(8),
                                                F(4), nullptr, dA1, synP, memP, H2, dmpS, dmpM, H2,
                                                membuf, spkbuf, bar, S);
  gemm(dA1, pWih_d1, F(17), xg, B * S, 4 * H2, H2);
  (void)hipMemsetAsync(bar, 0, 64, stream);
  k_slstm_scan<H2><<<H2 / 64, 128, 0, stream>>>(xg, pWhh_d1, F(16), pWfc_d1, F(15), F(19), F(13),
                                                F(18), F(14), dA1, dA2, nullptr, nullptr, H2, dmpS,
                                                dmpM, H2, membuf, spkbuf, bar, S);
  gemm(dA2, pOutW, F(63), logits, B * S, V, H2);
}
